// Qriaffine_72275709657290
// MI455X (gfx1250) — matrix operand order repaired and checked by reading, not compiled, not run on hardware
//
#include <hip/hip_runtime.h>
#include <hip/hip_bf16.h>

// ---------------------------------------------------------------------------
// Qriaffine on MI455X (gfx1250, wave32, WMMA bf16):
//   P[r,(a,c)] = l1[r,a]*h1[r,c]      (8192 x 4225, padded to 4352)
//   Wp[(a,c),(b,d)] = W[a,b,c,d]      (4225 x 4225, padded to 4352 x 4352)
//   Q = P @ Wp                        (8192 x 4352 x 4352 bf16 GEMM, f32 acc)
//   out[n] = Q[n] @ R[n]^T            (8 x (1024 x 1024 x 4352))
// LDS tiles live in WMMA-fragment order (fragment = one 32B ds_load_b128
// pair per lane). Staging uses GLOBAL_LOAD_ASYNC_TO_LDS_B128 with DOUBLE
// BUFFERING: tile k+1 streams into the alternate LDS buffer while the 8
// WMMAs of tile k execute; one barrier + one s_wait_asynccnt per K-step.
// ---------------------------------------------------------------------------

typedef __attribute__((ext_vector_type(16))) __bf16 v16bf;
typedef __attribute__((ext_vector_type(8)))  float  v8f;

#define DP    4352      // padded d*d: 65*65=4225 -> 34*128
#define NROWS 8192      // n * L
#define TM    128
#define TN    128
#define TK    32
#define TILE_ELEMS 4096         // 8 tiles * 32 lanes * 16 elems
#define BUF_BYTES  8192         // TILE_ELEMS * sizeof(bf16)

__device__ __forceinline__ void async_ld_b128(unsigned lds_off, const void* g)
{
  asm volatile("global_load_async_to_lds_b128 %0, %1, off"
               :: "v"(lds_off), "v"(g) : "memory");
}

__device__ __forceinline__ void wait_async0()
{
#if __has_builtin(__builtin_amdgcn_s_wait_asynccnt)
  __builtin_amdgcn_s_wait_asynccnt(0);
#else
  asm volatile("s_wait_asynccnt 0x0" ::: "memory");
#endif
}

// --- pack P (or R): out[r, a*65+c] = (a<64 ? lay[r,a] : 1) * h[r,c] ---------
__global__ __launch_bounds__(256) void qri_pack_pr(
    const float* __restrict__ lay, const float* __restrict__ h,
    __bf16* __restrict__ out)
{
  __shared__ float sl[65];
  __shared__ float sh[65];
  const int r = blockIdx.x;
  const int t = threadIdx.x;
  if (t < 64) sl[t] = lay[(size_t)r * 64 + t];
  if (t == 64) sl[64] = 1.0f;
  if (t < 65) sh[t] = h[(size_t)r * 65 + t];
  __syncthreads();
  __bf16* orow = out + (size_t)r * DP;
  #pragma unroll
  for (int it = 0; it < DP / 256; ++it) {
    int col = t + it * 256;
    float v = 0.0f;
    if (col < 4225) {
      int a = col / 65;
      int c = col - a * 65;
      v = sl[a] * sh[c];
    }
    orow[col] = (__bf16)v;
  }
}

// --- pack W: Wp[a*65+c, b*65+d] = W[a,b,c,d], zero padding -------------------
__global__ __launch_bounds__(256) void qri_pack_w(
    const float* __restrict__ W, __bf16* __restrict__ out)
{
  const int row = blockIdx.x;            // (a,c) index, 0..DP-1
  const int t   = threadIdx.x;
  const int a = row / 65;
  const int c = row - a * 65;
  const bool rok = (row < 4225);
  __bf16* orow = out + (size_t)row * DP;
  #pragma unroll
  for (int it = 0; it < DP / 256; ++it) {
    int col = t + it * 256;
    float v = 0.0f;
    if (rok && col < 4225) {
      int b = col / 65;
      int d = col - b * 65;
      v = W[(((size_t)a * 65 + b) * 65 + c) * 65 + d];
    }
    orow[col] = (__bf16)v;
  }
}

// --- bf16 WMMA GEMM: C = A(MxK) * B, BT: B is [N,ldb] (use B^T), else [K,ldb]
// 256 threads = 8 wave32; tile 128x128; each wave owns 32(M) x 64(N).
// LDS layout (fragment order): elem(tile, lane, e) = tile*512 + lane*16 + e.
template <bool BT, bool OUT_BF16>
__global__ __launch_bounds__(256) void qri_gemm(
    const __bf16* __restrict__ A, const __bf16* __restrict__ B,
    void* __restrict__ C,
    int lda, int ldb, int ldc, int K,
    long long strideA, long long strideB, long long strideC)
{
  __shared__ __bf16 sA[2][TILE_ELEMS];
  __shared__ __bf16 sB[2][TILE_ELEMS];

  const int t     = threadIdx.x;
  const int lane  = t & 31;
  const int wave  = t >> 5;
  const int waveM = wave & 3;        // 4 waves along M: 32 rows each
  const int waveN = wave >> 2;       // 2 waves along N: 64 cols each
  const int lrow  = lane & 15;
  const int hi    = lane >> 4;

  const long long z = blockIdx.z;
  A += z * strideA;
  B += z * strideB;
  const int m0 = blockIdx.y * TM;
  const int n0 = blockIdx.x * TN;

  const __bf16* Abase = A + (size_t)m0 * lda;
  const __bf16* Bbase = BT ? (B + (size_t)n0 * ldb) : (B + n0);

  // ---- loop-invariant async-staging slots: thread -> two 16B chunks --------
  // chunk p -> (row, g): k = 8g; A frag dest lane = (row&15) + ((g>=2)<<4),
  // e-base = (g&1)*8; NT B frag dest lane = (c&15) + ((g>=2)<<4),
  // e-base = (g&1)*8. Both keep the 16B chunk contiguous in LDS.
  unsigned aoff[2], boff[2];
  int srow[2], sk[2];
  #pragma unroll
  for (int i = 0; i < 2; ++i) {
    int p  = i * 256 + t;            // 0..511
    int rw = p >> 2;                 // 0..127
    int g  = p & 3;                  // 16B chunk within 32 k's
    srow[i] = rw;
    sk[i]   = g << 3;
    int tile = rw >> 4;
    aoff[i] = (unsigned)(size_t)(
        &sA[0][tile * 512 + ((rw & 15) + ((g >> 1) << 4)) * 16 +
               ((g & 1) << 3)]);
    boff[i] = (unsigned)(size_t)(
        &sB[0][tile * 512 + ((rw & 15) + ((g >> 1) << 4)) * 16 +
               ((g & 1) << 3)]);
  }

  // ---- staging (issue only; completion via ASYNCcnt / DScnt + barrier) -----
  auto stage = [&](int kks, int buf) {
    unsigned bo = (unsigned)buf * BUF_BYTES;
    #pragma unroll
    for (int i = 0; i < 2; ++i)
      async_ld_b128(aoff[i] + bo, Abase + (size_t)srow[i] * lda + kks + sk[i]);
    if (BT) {
      #pragma unroll
      for (int i = 0; i < 2; ++i)
        async_ld_b128(boff[i] + bo,
                      Bbase + (size_t)srow[i] * ldb + kks + sk[i]);
    } else {
      // Wp rows: N contiguous; load 8 cols (uint4), scatter 8 ds_store_b16
      __bf16* dst = &sB[buf][0];
      #pragma unroll
      for (int i = 0; i < 2; ++i) {
        int p    = i * 256 + t;               // 0..511
        int k    = p >> 4;                    // 0..31
        int c8   = (p & 15) << 3;             // 0..120 step 8
        uint4 v  = *(const uint4*)(Bbase + (size_t)(kks + k) * ldb + c8);
        const __bf16* pv = (const __bf16*)&v;
        int e    = k & 15;
        int hb   = (k >> 4) << 4;
        #pragma unroll
        for (int j = 0; j < 8; ++j) {
          int col = c8 + j;
          dst[(col >> 4) * 512 + ((col & 15) + hb) * 16 + e] = pv[j];
        }
      }
    }
  };

  v8f acc[2][4];
  #pragma unroll
  for (int i = 0; i < 2; ++i)
    #pragma unroll
    for (int j = 0; j < 4; ++j)
      #pragma unroll
      for (int r = 0; r < 8; ++r) acc[i][j][r] = 0.0f;

  int buf = 0;
  stage(0, 0);                        // prologue: fill buffer 0

  for (int kk = 0; kk < K; kk += TK) {
    wait_async0();                    // my DMAs into buf landed
    __syncthreads();                  // everyone's DMAs/stores visible; prior
                                      // reads of buf^1 are complete (dscnt)
    if (kk + TK < K) {
      if (!BT)
        __builtin_prefetch(Bbase + (size_t)(kk + 2 * TK + (t >> 3)) * ldb +
                               ((t & 7) << 4), 0, 1);
      stage(kk + TK, buf ^ 1);        // overlap next tile with WMMAs below
    }

    // ---- fragments: one aligned 32B read per tile per lane ----------------
    v16bf afr[2];
    #pragma unroll
    for (int tm = 0; tm < 2; ++tm)
      afr[tm] =
          *(const v16bf*)(&sA[buf][(waveM * 2 + tm) * 512 + lane * 16]);
    v16bf bfr[4];
    #pragma unroll
    for (int tn = 0; tn < 4; ++tn)
      bfr[tn] =
          *(const v16bf*)(&sB[buf][(waveN * 4 + tn) * 512 + lane * 16]);

    #pragma unroll
    for (int tm = 0; tm < 2; ++tm)
      #pragma unroll
      for (int tn = 0; tn < 4; ++tn)
        acc[tm][tn] = __builtin_amdgcn_wmma_f32_16x16x32_bf16(
            false, afr[tm], false, bfr[tn], (short)0, acc[tm][tn],
            false, false);
    buf ^= 1;
  }

  // ---- store: C/D layout: VGPR r -> M = r + (lane>=16 ? 8 : 0), N = lane&15
  const int rofs = hi * 8;
  #pragma unroll
  for (int tm = 0; tm < 2; ++tm)
    #pragma unroll
    for (int tn = 0; tn < 4; ++tn) {
      int nn = n0 + waveN * 64 + tn * 16 + lrow;
      #pragma unroll
      for (int r = 0; r < 8; ++r) {
        int mm = m0 + waveM * 32 + tm * 16 + rofs + r;
        if (OUT_BF16)
          ((__bf16*)C)[z * strideC + (size_t)mm * ldc + nn] =
              (__bf16)acc[tm][tn][r];
        else
          ((float*)C)[z * strideC + (size_t)mm * ldc + nn] = acc[tm][tn][r];
      }
    }
}

extern "C" void kernel_launch(void* const* d_in, const int* in_sizes, int n_in,
                              void* d_out, int out_size, void* d_ws, size_t ws_size,
                              hipStream_t stream)
{
  const float* layer1 = (const float*)d_in[0];
  const float* layer2 = (const float*)d_in[1];
  const float* h1     = (const float*)d_in[3];
  const float* h2     = (const float*)d_in[4];
  const float* W      = (const float*)d_in[6];
  float* out          = (float*)d_out;

  // workspace: P, R (8192 x 4352), Wp (4352 x 4352), Q (8192 x 4352), bf16
  __bf16* P  = (__bf16*)d_ws;
  __bf16* R  = P  + (size_t)NROWS * DP;
  __bf16* Wp = R  + (size_t)NROWS * DP;
  __bf16* Q  = Wp + (size_t)DP * DP;

  qri_pack_pr<<<NROWS, 256, 0, stream>>>(layer1, h1, P);
  qri_pack_pr<<<NROWS, 256, 0, stream>>>(layer2, h2, R);
  qri_pack_w <<<DP,    256, 0, stream>>>(W, Wp);

  // Q = P @ Wp   (NN, bf16 output)
  qri_gemm<false, true><<<dim3(DP / TN, NROWS / TM, 1), 256, 0, stream>>>(
      P, Wp, Q, DP, DP, DP, DP, 0LL, 0LL, 0LL);

  // out[n] = Q[n] @ R[n]^T   (NT, f32 output), batched over n=8
  qri_gemm<true, false><<<dim3(1024 / TN, 1024 / TM, 8), 256, 0, stream>>>(
      Q, R, out, DP, DP, 1024, DP,
      (long long)1024 * DP, (long long)1024 * DP, (long long)1024 * 1024);
}
